// PatchNet_12884901888412
// MI455X (gfx1250) — matrix operand order repaired and checked by reading, not compiled, not run on hardware
//
#include <hip/hip_runtime.h>
#include <hip/hip_bf16.h>

// ---------------------------------------------------------------------------
// CDNA5 / gfx1250 DGCNN-style PatchNet forward.
// wave32 WMMA 16x16 tiles: bf16 WMMA (f32 accum) for all channel GEMMs,
// f32 WMMA 16x16x4 for exact pairwise distances feeding top-k.
// All operand construction is branch-free (clamped addresses + selects) so
// EXEC stays all-ones around every WMMA.
// ---------------------------------------------------------------------------

typedef __bf16 bf16_t;
typedef bf16_t v16bf __attribute__((ext_vector_type(16)));
typedef float  v8f   __attribute__((ext_vector_type(8)));
typedef float  v2f   __attribute__((ext_vector_type(2)));

#define BATCH   16
#define NPTS    1024
#define KNN_K   20
#define KNN_KMX 80
#define NEG_INF (-3.402823466e38f)

static inline int cdiv(int a, int b) { return (a + b - 1) / b; }

__device__ __forceinline__ float lrelu(float x) { return x > 0.f ? x : 0.2f * x; }
__device__ __forceinline__ float bn_one(float x, float g, float b, float rm, float rv) {
  return (x - rm) * __frsqrt_rn(rv + 1e-5f) * g + b;
}
__device__ __forceinline__ bf16_t f2bf(float f) {
  unsigned u = __builtin_bit_cast(unsigned, f);
  unsigned r = (u + 0x7FFFu + ((u >> 16) & 1u)) >> 16;  // RNE f32->bf16
  unsigned short s = (unsigned short)r;
  return __builtin_bit_cast(bf16_t, s);
}
__device__ __forceinline__ v8f wmma_bf16(v16bf a, v16bf b, v8f c) {
  return __builtin_amdgcn_wmma_f32_16x16x32_bf16(false, a, false, b, (short)0, c,
                                                 false, false);
}
// Load one 16x32 bf16 A fragment (weights, row-major [Cout][CIN]) with b128s.
template <int CIN>
__device__ __forceinline__ v16bf load_A_frag(const float* __restrict__ W, int mc,
                                             int c0, int half) {
  const float* wp = W + (long)mc * CIN + c0 + 16 * half;
  float4 w0 = *(const float4*)(wp + 0);
  float4 w1 = *(const float4*)(wp + 4);
  float4 w2 = *(const float4*)(wp + 8);
  float4 w3 = *(const float4*)(wp + 12);
  v16bf A;
  A[0] = f2bf(w0.x);  A[1] = f2bf(w0.y);  A[2] = f2bf(w0.z);  A[3] = f2bf(w0.w);
  A[4] = f2bf(w1.x);  A[5] = f2bf(w1.y);  A[6] = f2bf(w1.z);  A[7] = f2bf(w1.w);
  A[8] = f2bf(w2.x);  A[9] = f2bf(w2.y);  A[10] = f2bf(w2.z); A[11] = f2bf(w2.w);
  A[12] = f2bf(w3.x); A[13] = f2bf(w3.y); A[14] = f2bf(w3.z); A[15] = f2bf(w3.w);
  return A;
}

// ---------------------------------------------------------------------------
// glue kernels
// ---------------------------------------------------------------------------

// pointcloud [B][N][3] -> xyzT [B][3][N]
__global__ void k_transpose(const float* __restrict__ pc, float* __restrict__ xyzT) {
  int i = blockIdx.x * blockDim.x + threadIdx.x;
  if (i >= BATCH * NPTS) return;
  int b = i / NPTS, n = i % NPTS;
#pragma unroll
  for (int c = 0; c < 3; ++c)
    xyzT[((long)b * 3 + c) * NPTS + n] = pc[(long)i * 3 + c];
}

// [B][C][N] -> point-major padded [B][N][CPAD] (pad channels = 0)
template <int C, int CPAD>
__global__ void k_featT(const float* __restrict__ X, float* __restrict__ XT) {
  int i = blockIdx.x * blockDim.x + threadIdx.x;
  if (i >= BATCH * NPTS) return;
  int b = i / NPTS, n = i % NPTS;
  const float* Xb = X + (long)b * C * NPTS + n;
  float* o = XT + (long)i * CPAD;
#pragma unroll
  for (int c = 0; c < CPAD; ++c) o[c] = (c < C) ? Xb[(long)c * NPTS] : 0.f;
}

// xx[b][n] = sum_c XT[b][n][c]^2   (float4 reads, pad contributes 0)
template <int CPAD>
__global__ void k_xx_t(const float* __restrict__ XT, float* __restrict__ xx) {
  int i = blockIdx.x * blockDim.x + threadIdx.x;
  if (i >= BATCH * NPTS) return;
  const float4* p = (const float4*)(XT + (long)i * CPAD);
  float s = 0.f;
#pragma unroll
  for (int c = 0; c < CPAD / 4; ++c) {
    float4 v = p[c];
    s += v.x * v.x + v.y * v.y + v.z * v.z + v.w * v.w;
  }
  xx[i] = s;
}

// quaternion [4][16] (+[1,0,0,0]) -> rotation matrices
__global__ void k_quat(const float* __restrict__ q4, float* __restrict__ trans_out,
                       float* __restrict__ Rws) {
  int b = threadIdx.x;
  if (b >= BATCH) return;
  float w = q4[0 * BATCH + b] + 1.f;
  float x = q4[1 * BATCH + b];
  float y = q4[2 * BATCH + b];
  float z = q4[3 * BATCH + b];
  float s = 2.f / (w * w + x * x + y * y + z * z);
  float R[9] = {1.f - s * (y * y + z * z), s * (x * y - z * w), s * (x * z + y * w),
                s * (x * y + z * w), 1.f - s * (x * x + z * z), s * (y * z - x * w),
                s * (x * z - y * w), s * (y * z + x * w), 1.f - s * (x * x + y * y)};
#pragma unroll
  for (int i = 0; i < 9; ++i) { trans_out[b * 9 + i] = R[i]; Rws[b * 9 + i] = R[i]; }
}

// xyz_rot[b][d][n] = sum_c xyzT[b][c][n] * R[b][c][d]
__global__ void k_rotate(const float* __restrict__ xyzT, const float* __restrict__ R,
                         float* __restrict__ xyzrot, float* __restrict__ out_xyz) {
  int i = blockIdx.x * blockDim.x + threadIdx.x;
  if (i >= BATCH * NPTS) return;
  int b = i / NPTS, n = i % NPTS;
  const float* Rb = R + b * 9;
  float v0 = xyzT[((long)b * 3 + 0) * NPTS + n];
  float v1 = xyzT[((long)b * 3 + 1) * NPTS + n];
  float v2 = xyzT[((long)b * 3 + 2) * NPTS + n];
#pragma unroll
  for (int d = 0; d < 3; ++d) {
    float s = v0 * Rb[0 * 3 + d] + v1 * Rb[1 * 3 + d] + v2 * Rb[2 * 3 + d];
    xyzrot[((long)b * 3 + d) * NPTS + n] = s;
    out_xyz[((long)b * 3 + d) * NPTS + n] = s;
  }
}

// max over N: X [B][C][N] -> out [C][B]  (so linears see [Cin][P=16])
__global__ void k_rmaxN(const float* __restrict__ X, float* __restrict__ out, int C) {
  int i = blockIdx.x * blockDim.x + threadIdx.x;
  if (i >= BATCH * C) return;
  int b = i / C, c = i % C;
  const float4* p = (const float4*)(X + ((long)b * C + c) * NPTS);
  float m = NEG_INF;
  for (int n = 0; n < NPTS / 4; ++n) {
    float4 v = p[n];
    m = fmaxf(m, fmaxf(fmaxf(v.x, v.y), fmaxf(v.z, v.w)));
  }
  out[c * BATCH + b] = m;
}

// concat [B][256][N] with broadcast global [256][B] -> [B][512][N]
__global__ void k_concat(const float* __restrict__ h, const float* __restrict__ gmx,
                         float* __restrict__ out) {
  int i = blockIdx.x * blockDim.x + threadIdx.x;
  if (i >= BATCH * 512 * NPTS) return;
  int b = i / (512 * NPTS);
  int r = i % (512 * NPTS);
  int c = r / NPTS, n = r % NPTS;
  out[i] = (c < 256) ? h[((long)b * 256 + c) * NPTS + n] : gmx[(c - 256) * BATCH + b];
}

// ---------------------------------------------------------------------------
// pairwise-distance GEMM from point-major XT [B][N][CPAD]:
//   pd[b][n][m] = 2*<x_n,x_m> - xx[n] - xx[m]
// One f32 WMMA per 4 channels; each lane's fragment is a single b64 load.
// ---------------------------------------------------------------------------
#if __has_builtin(__builtin_amdgcn_wmma_f32_16x16x4_f32)
#define PD_WMMA 1
#endif

template <int CPAD>
__global__ __launch_bounds__(128) void k_pd_t(const float* __restrict__ XT,
                                              const float* __restrict__ xx,
                                              float* __restrict__ pd) {
  int lane = threadIdx.x & 31, wave = threadIdx.x >> 5;
  int half = lane >> 4, l15 = lane & 15;
  int m0 = (blockIdx.x * 4 + wave) * 16;
  int n0 = blockIdx.y * 16;
  int b  = blockIdx.z;
  const float* Tb  = XT + (long)b * NPTS * CPAD;
  const float* xxb = xx + (long)b * NPTS;
  const float* an = Tb + (long)(n0 + l15) * CPAD + 2 * half;
  const float* bm = Tb + (long)(m0 + l15) * CPAD + 2 * half;
  v8f acc = {};
#ifdef PD_WMMA
#pragma unroll
  for (int c0 = 0; c0 < CPAD; c0 += 4) {
    v2f av = *(const v2f*)(an + c0);
    v2f bv = *(const v2f*)(bm + c0);
    acc = __builtin_amdgcn_wmma_f32_16x16x4_f32(false, av, false, bv, (short)0, acc,
                                                false, false);
  }
#else
#pragma unroll
  for (int v = 0; v < 8; ++v) {
    int n = n0 + v + half * 8, m = m0 + l15;
    float s = 0.f;
    for (int c = 0; c < CPAD; ++c) s += Tb[(long)n * CPAD + c] * Tb[(long)m * CPAD + c];
    acc[v] = s;
  }
#endif
#pragma unroll
  for (int v = 0; v < 8; ++v) {
    int n = n0 + v + half * 8, m = m0 + l15;
    pd[((long)b * NPTS + n) * NPTS + m] = 2.f * acc[v] - xxb[n] - xxb[m];
  }
}

// top-k per row: one wave per row; row staged in LDS; wave argmax via shfl_xor
__global__ __launch_bounds__(128) void k_topk(const float* __restrict__ pd,
                                              int* __restrict__ idx, int kk) {
  __shared__ float row[4][NPTS];
  int lane = threadIdx.x & 31, wave = threadIdx.x >> 5;
  int n = blockIdx.x * 4 + wave;
  int b = blockIdx.y;
  const float* src = pd + ((long)b * NPTS + n) * NPTS;
  for (int i = lane; i < NPTS; i += 32) row[wave][i] = src[i];
  for (int t = 0; t < kk; ++t) {
    float best = NEG_INF; int bi = NPTS;
    for (int i = lane; i < NPTS; i += 32) {
      float v = row[wave][i];
      if (v > best || (v == best && i < bi)) { best = v; bi = i; }
    }
#pragma unroll
    for (int off = 16; off > 0; off >>= 1) {
      float ov = __shfl_xor(best, off, 32);
      int   oi = __shfl_xor(bi, off, 32);
      if (ov > best || (ov == best && oi < bi)) { best = ov; bi = oi; }
    }
    if (lane == 0) {
      idx[((long)b * NPTS + n) * KNN_KMX + t] = bi;
      row[wave][bi] = NEG_INF;
    }
  }
}

// ---------------------------------------------------------------------------
// generic bf16 WMMA GEMM (K templated): Y = act(bn(W @ X + bias))
// X:[CIN][P], W:[Cout][CIN]; one wave -> 16(out) x 16(P) tile.
// flags: bit0 = BN, bit1 = LeakyReLU. P must be a multiple of 16.
// ---------------------------------------------------------------------------
template <int CIN>
__global__ __launch_bounds__(128) void k_gemm_t(
    const float* __restrict__ X, const float* __restrict__ W,
    const float* __restrict__ bias, const float* __restrict__ g,
    const float* __restrict__ bb, const float* __restrict__ rm,
    const float* __restrict__ rv, float* __restrict__ Y, int Cout, int P,
    long strideXb, long strideYb, int flags) {
  int lane = threadIdx.x & 31, wave = threadIdx.x >> 5;
  int half = lane >> 4, l15 = lane & 15;
  int p0 = (blockIdx.x * 4 + wave) * 16;
  int m0 = blockIdx.y * 16;
  int b  = blockIdx.z;
  if (p0 >= P) return;
  const float* Xb = X + (long)b * strideXb;
  float* Yb = Y + (long)b * strideYb;
  int m  = m0 + l15;
  int mc = m < Cout ? m : Cout - 1;
  int p  = p0 + l15;
  v8f acc = {};
  for (int c0 = 0; c0 < CIN; c0 += 32) {
    v16bf A = load_A_frag<CIN>(W, mc, c0, half);
    const float* xp = Xb + (long)(c0 + 16 * half) * P + p;
    if (c0 + 32 < CIN) __builtin_prefetch(xp + 32l * P, 0, 1);
    v16bf B;
#pragma unroll
    for (int h = 0; h < 16; ++h) B[h] = f2bf(xp[(long)h * P]);
    acc = wmma_bf16(A, B, acc);
  }
#pragma unroll
  for (int v = 0; v < 8; ++v) {
    int mm = m0 + v + half * 8;
    if (mm < Cout) {
      float y = acc[v] + bias[mm];
      if (flags & 1) y = bn_one(y, g[mm], bb[mm], rm[mm], rv[mm]);
      if (flags & 2) y = lrelu(y);
      Yb[(long)mm * P + p] = y;
    }
  }
}

// ---------------------------------------------------------------------------
// gather-conv over (n,j), Cin = 6 (xyz edge features), no max:
// out [B][Cout][N*kk], BN + LeakyReLU. Branch-free gather via clamped indices.
// ---------------------------------------------------------------------------
__global__ __launch_bounds__(128) void k_gconv6(
    const float* __restrict__ X, const int* __restrict__ idx,
    const float* __restrict__ W, const float* __restrict__ bias,
    const float* __restrict__ g, const float* __restrict__ bb,
    const float* __restrict__ rm, const float* __restrict__ rv,
    float* __restrict__ Y, int Cout, int kk) {
  constexpr int C = 3, CIN = 6;
  int lane = threadIdx.x & 31, wave = threadIdx.x >> 5;
  int half = lane >> 4, l15 = lane & 15;
  int P  = NPTS * kk;
  int q0 = (blockIdx.x * 4 + wave) * 16;
  int m0 = blockIdx.y * 16;
  int b  = blockIdx.z;
  if (q0 >= P) return;
  const float* Xb = X + (long)b * C * NPTS;
  const int* idxb = idx + (long)b * NPTS * KNN_KMX;
  int q = q0 + l15;
  int n = q / kk, j = q - n * kk;
  int mi = idxb[n * KNN_KMX + j];
  int m  = m0 + l15;
  int mc = m < Cout ? m : Cout - 1;
  v16bf A;
#pragma unroll
  for (int h = 0; h < 16; ++h) {
    int ka = h + 16 * half;
    float wv = W[(long)mc * CIN + (ka < CIN ? ka : CIN - 1)];
    A[h] = f2bf(ka < CIN ? wv : 0.f);
  }
  v16bf B;
#pragma unroll
  for (int h = 0; h < 16; ++h) {
    int kb = h + 16 * half;
    int cc = kb < C ? kb : (kb < CIN ? kb - C : 0);
    float ctr = Xb[(long)cc * NPTS + n];
    float ngh = Xb[(long)(kb < C ? kb : C - 1) * NPTS + mi];
    float xv  = kb < C ? (ngh - ctr) : (kb < CIN ? ctr : 0.f);
    B[h] = f2bf(xv);
  }
  v8f acc = {};
  acc = wmma_bf16(A, B, acc);
  float* Yb = Y + (long)b * Cout * (long)P;
#pragma unroll
  for (int v = 0; v < 8; ++v) {
    int mm = m0 + v + half * 8;
    if (mm < Cout) {
      float y = lrelu(bn_one(acc[v] + bias[mm], g[mm], bb[mm], rm[mm], rv[mm]));
      Yb[(long)mm * P + q] = y;
    }
  }
}

// ---------------------------------------------------------------------------
// GEMM over materialized [CIN][N*kk] with fused max over j -> [Cout][N]
// ---------------------------------------------------------------------------
template <int CIN>
__global__ __launch_bounds__(128) void k_gemm_maxk_t(
    const float* __restrict__ X, const float* __restrict__ W,
    const float* __restrict__ bias, const float* __restrict__ g,
    const float* __restrict__ bb, const float* __restrict__ rm,
    const float* __restrict__ rv, float* __restrict__ Y, int Cout, int kk) {
  constexpr int NCH = CIN / 32;
  int lane = threadIdx.x & 31, wave = threadIdx.x >> 5;
  int half = lane >> 4, l15 = lane & 15;
  int n0 = (blockIdx.x * 4 + wave) * 16;
  int m0 = blockIdx.y * 16;
  int b  = blockIdx.z;
  long Pin = (long)NPTS * kk;
  const float* Xb = X + (long)b * CIN * Pin;
  int n  = n0 + l15;
  int m  = m0 + l15;
  int mc = m < Cout ? m : Cout - 1;
  v16bf A[NCH];
#pragma unroll
  for (int ch = 0; ch < NCH; ++ch) A[ch] = load_A_frag<CIN>(W, mc, ch * 32, half);
  float maxv[8];
#pragma unroll
  for (int v = 0; v < 8; ++v) maxv[v] = NEG_INF;
  for (int j = 0; j < kk; ++j) {
    v8f acc = {};
#pragma unroll
    for (int ch = 0; ch < NCH; ++ch) {
      const float* xp = Xb + (long)(ch * 32 + 16 * half) * Pin + (long)n * kk + j;
      v16bf B;
#pragma unroll
      for (int h = 0; h < 16; ++h) B[h] = f2bf(xp[(long)h * Pin]);
      acc = wmma_bf16(A[ch], B, acc);
    }
#pragma unroll
    for (int v = 0; v < 8; ++v) {
      int mm = m0 + v + half * 8;
      float y = acc[v] + bias[mm < Cout ? mm : Cout - 1];
      if (mm < Cout) {
        y = lrelu(bn_one(y, g[mm], bb[mm], rm[mm], rv[mm]));
        maxv[v] = fmaxf(maxv[v], y);
      }
    }
  }
  float* Yb = Y + (long)b * Cout * NPTS;
#pragma unroll
  for (int v = 0; v < 8; ++v) {
    int mm = m0 + v + half * 8;
    if (mm < Cout) Yb[(long)mm * NPTS + n] = maxv[v];
  }
}

// ---------------------------------------------------------------------------
// fused edge-conv: gather(2C edge feats) -> WMMA -> BN -> LeakyReLU -> max_j
// (+ optional residual). idx stride d implements knn(k*d)[::d].
// For C % 32 == 0: lower C channels are gather-diff chunks, upper C channels
// (center features) are j-invariant -> precomputed bf16 fragments.
// ---------------------------------------------------------------------------
template <int C>
__global__ __launch_bounds__(128) void k_edgeconv_t(
    const float* __restrict__ X, const int* __restrict__ idx,
    const float* __restrict__ W, const float* __restrict__ bias,
    const float* __restrict__ g, const float* __restrict__ bb,
    const float* __restrict__ rm, const float* __restrict__ rv,
    const float* __restrict__ resid, float* __restrict__ Y, int Cout, int kk,
    int d) {
  constexpr int CIN = 2 * C;
  constexpr int NCH = (CIN + 31) / 32;
  int lane = threadIdx.x & 31, wave = threadIdx.x >> 5;
  int half = lane >> 4, l15 = lane & 15;
  int n0 = (blockIdx.x * 4 + wave) * 16;
  int m0 = blockIdx.y * 16;
  int b  = blockIdx.z;
  const float* Xb = X + (long)b * C * NPTS;
  const int* idxb = idx + (long)b * NPTS * KNN_KMX;
  int n  = n0 + l15;
  int m  = m0 + l15;
  int mc = m < Cout ? m : Cout - 1;
  v16bf A[NCH];
  if constexpr (CIN % 32 == 0) {
#pragma unroll
    for (int ch = 0; ch < NCH; ++ch) A[ch] = load_A_frag<CIN>(W, mc, ch * 32, half);
  } else {
#pragma unroll
    for (int ch = 0; ch < NCH; ++ch)
#pragma unroll
      for (int h = 0; h < 16; ++h) {
        int ka = ch * 32 + h + 16 * half;
        float wv = W[(long)mc * CIN + (ka < CIN ? ka : CIN - 1)];
        A[ch][h] = f2bf(ka < CIN ? wv : 0.f);
      }
  }
  float maxv[8];
#pragma unroll
  for (int v = 0; v < 8; ++v) maxv[v] = NEG_INF;

  if constexpr (C % 32 == 0) {
    constexpr int ND = C / 32;  // gather-diff chunks
    float ctrD[ND][16];
    v16bf Bct[NCH - ND];
#pragma unroll
    for (int ch = 0; ch < ND; ++ch)
#pragma unroll
      for (int h = 0; h < 16; ++h)
        ctrD[ch][h] = Xb[(long)(ch * 32 + h + 16 * half) * NPTS + n];
#pragma unroll
    for (int ch = ND; ch < NCH; ++ch)
#pragma unroll
      for (int h = 0; h < 16; ++h)
        Bct[ch - ND][h] = f2bf(Xb[(long)(ch * 32 + h + 16 * half - C) * NPTS + n]);
    for (int j = 0; j < kk; ++j) {
      int mi = idxb[n * KNN_KMX + j * d];
      v8f acc = {};
#pragma unroll
      for (int ch = 0; ch < ND; ++ch) {
        const float* xp = Xb + (long)(ch * 32 + 16 * half) * NPTS + mi;
        v16bf B;
#pragma unroll
        for (int h = 0; h < 16; ++h) B[h] = f2bf(xp[(long)h * NPTS] - ctrD[ch][h]);
        acc = wmma_bf16(A[ch], B, acc);
      }
#pragma unroll
      for (int ch = ND; ch < NCH; ++ch) acc = wmma_bf16(A[ch], Bct[ch - ND], acc);
#pragma unroll
      for (int v = 0; v < 8; ++v) {
        int mm = m0 + v + half * 8;
        if (mm < Cout) {
          float y = lrelu(bn_one(acc[v] + bias[mm], g[mm], bb[mm], rm[mm], rv[mm]));
          maxv[v] = fmaxf(maxv[v], y);
        }
      }
    }
  } else {
    // small C (=3): single mixed chunk, branch-free selects
    float ctrA[16];
#pragma unroll
    for (int h = 0; h < 16; ++h) {
      int kb = h + 16 * half;
      int cc = kb < C ? kb : (kb < CIN ? kb - C : 0);
      ctrA[h] = (kb < CIN) ? Xb[(long)cc * NPTS + n] : 0.f;
    }
    for (int j = 0; j < kk; ++j) {
      int mi = idxb[n * KNN_KMX + j * d];
      v16bf B;
#pragma unroll
      for (int h = 0; h < 16; ++h) {
        int kb = h + 16 * half;
        float ngh = Xb[(long)(kb < C ? kb : C - 1) * NPTS + mi];
        float xv  = kb < C ? (ngh - ctrA[h]) : ctrA[h];
        B[h] = f2bf(xv);
      }
      v8f acc = {};
      acc = wmma_bf16(A[0], B, acc);
#pragma unroll
      for (int v = 0; v < 8; ++v) {
        int mm = m0 + v + half * 8;
        if (mm < Cout) {
          float y = lrelu(bn_one(acc[v] + bias[mm], g[mm], bb[mm], rm[mm], rv[mm]));
          maxv[v] = fmaxf(maxv[v], y);
        }
      }
    }
  }
  float* Yb = Y + (long)b * Cout * NPTS;
#pragma unroll
  for (int v = 0; v < 8; ++v) {
    int mm = m0 + v + half * 8;
    if (mm < Cout) {
      float y = maxv[v];
      if (resid) y += resid[((long)b * Cout + mm) * NPTS + n];
      Yb[(long)mm * NPTS + n] = y;
    }
  }
}

// ---------------------------------------------------------------------------
// host orchestration
// ---------------------------------------------------------------------------
extern "C" void kernel_launch(void* const* d_in, const int* in_sizes, int n_in,
                              void* d_out, int out_size, void* d_ws, size_t ws_size,
                              hipStream_t stream) {
  (void)in_sizes; (void)out_size;
  if (n_in < 89) return;
  const float* pc = (const float*)d_in[0];
#define IN(i) ((const float*)d_in[i])
  // params pytree, keys sorted: bn1..bn4, c1..c4, sn, tn ; bn:{b,g,rm,rv}; conv:{bias,w}
  const float *M_bn1_b=IN(1),*M_bn1_g=IN(2),*M_bn1_rm=IN(3),*M_bn1_rv=IN(4);
  const float *M_bn2_b=IN(5),*M_bn2_g=IN(6),*M_bn2_rm=IN(7),*M_bn2_rv=IN(8);
  const float *M_bn3_b=IN(9),*M_bn3_g=IN(10),*M_bn3_rm=IN(11),*M_bn3_rv=IN(12);
  const float *M_bn4_b=IN(13),*M_bn4_g=IN(14),*M_bn4_rm=IN(15),*M_bn4_rv=IN(16);
  const float *M_c1_bias=IN(17),*M_c1_w=IN(18);
  const float *M_c2_bias=IN(19),*M_c2_w=IN(20);
  const float *M_c3_bias=IN(21),*M_c3_w=IN(22);
  const float *M_c4_bias=IN(23),*M_c4_w=IN(24);
  const float *S_bn1_b=IN(25),*S_bn1_g=IN(26),*S_bn1_rm=IN(27),*S_bn1_rv=IN(28);
  const float *S_bn2_b=IN(29),*S_bn2_g=IN(30),*S_bn2_rm=IN(31),*S_bn2_rv=IN(32);
  const float *S_bn3_b=IN(33),*S_bn3_g=IN(34),*S_bn3_rm=IN(35),*S_bn3_rv=IN(36);
  const float *S_bnd1_b=IN(37),*S_bnd1_g=IN(38),*S_bnd1_rm=IN(39),*S_bnd1_rv=IN(40);
  const float *S_bnd2_b=IN(41),*S_bnd2_g=IN(42),*S_bnd2_rm=IN(43),*S_bnd2_rv=IN(44);
  const float *S_c1_bias=IN(45),*S_c1_w=IN(46);
  const float *S_c2_bias=IN(47),*S_c2_w=IN(48);
  const float *S_c3_bias=IN(49),*S_c3_w=IN(50);
  const float *S_d1_bias=IN(51),*S_d1_w=IN(52);
  const float *S_d2_bias=IN(53),*S_d2_w=IN(54);
  const float *S_d3_bias=IN(55),*S_d3_w=IN(56);
  const float *T_bn1_b=IN(57),*T_bn1_g=IN(58),*T_bn1_rm=IN(59),*T_bn1_rv=IN(60);
  const float *T_bn2_b=IN(61),*T_bn2_g=IN(62),*T_bn2_rm=IN(63),*T_bn2_rv=IN(64);
  const float *T_bn3_b=IN(65),*T_bn3_g=IN(66),*T_bn3_rm=IN(67),*T_bn3_rv=IN(68);
  const float *T_bnl1_b=IN(69),*T_bnl1_g=IN(70),*T_bnl1_rm=IN(71),*T_bnl1_rv=IN(72);
  const float *T_bnl2_b=IN(73),*T_bnl2_g=IN(74),*T_bnl2_rm=IN(75),*T_bnl2_rv=IN(76);
  const float *T_c1_bias=IN(77),*T_c1_w=IN(78);
  const float *T_c2_bias=IN(79),*T_c2_w=IN(80);
  const float *T_c3_bias=IN(81),*T_c3_w=IN(82);
  const float *T_l1_bias=IN(83),*T_l1_w=IN(84);
  const float *T_l2_bias=IN(85),*T_l2_w=IN(86);
  const float *T_lt_bias=IN(87),*T_lt_w=IN(88);
#undef IN

  const long B64N = (long)BATCH * 64 * NPTS;
  float* out     = (float*)d_out;
  float* out_x2  = out;
  float* out_x3  = out + B64N;
  float* out_x4  = out + 2 * B64N;
  float* out_tr  = out + 3 * B64N;
  float* out_off = out_tr + BATCH * 9;
  float* out_xyz = out_off + (long)BATCH * 3 * NPTS;

  float* ws = (float*)d_ws;
  size_t o = 0;
  auto alloc = [&](size_t n) { float* p = ws + o; o += n; return p; };
  float* xyzT   = alloc((size_t)BATCH * 3 * NPTS);
  float* xyzrot = alloc((size_t)BATCH * 3 * NPTS);
  float* xxbuf  = alloc((size_t)BATCH * NPTS);
  float* Rbuf   = alloc((size_t)BATCH * 9);
  float* pooled = alloc((size_t)1024 * BATCH);
  float* l512   = alloc((size_t)512 * BATCH);
  float* l256   = alloc((size_t)256 * BATCH);
  float* q4     = alloc((size_t)4 * BATCH);
  float* g256   = alloc((size_t)256 * BATCH);
  float* xtbuf  = alloc((size_t)BATCH * NPTS * 64);   // point-major padded feats
  float* h128   = alloc((size_t)BATCH * 128 * NPTS);
  float* f256   = alloc((size_t)BATCH * 256 * NPTS);
  float* h64    = alloc((size_t)BATCH * 64 * NPTS);
  float* x1buf  = alloc((size_t)BATCH * 64 * NPTS);
  float* big1   = alloc((size_t)BATCH * 64 * NPTS * KNN_K);
  float* pdbuf  = alloc((size_t)BATCH * NPTS * NPTS);  // reused: tn.c3 out / sn concat
  int*   idxbuf = (int*)alloc((size_t)BATCH * NPTS * KNN_KMX);
  if (ws_size < o * sizeof(float)) return;

  const int TPB = 256;
  dim3 blk(128);
  const int BN = BATCH * NPTS;
  long sX64 = (long)64 * NPTS, sX128 = (long)128 * NPTS, sX256 = (long)256 * NPTS;
  long sX512 = (long)512 * NPTS, sY1024 = (long)1024 * NPTS;

  auto knn3 = [&](const float* feat, int kk) {
    k_featT<3, 4><<<cdiv(BN, TPB), TPB, 0, stream>>>(feat, xtbuf);
    k_xx_t<4><<<cdiv(BN, TPB), TPB, 0, stream>>>(xtbuf, xxbuf);
    k_pd_t<4><<<dim3(NPTS / 64, NPTS / 16, BATCH), blk, 0, stream>>>(xtbuf, xxbuf, pdbuf);
    k_topk<<<dim3(NPTS / 4, BATCH), blk, 0, stream>>>(pdbuf, idxbuf, kk);
  };
  auto knn64 = [&](const float* feat, int kk) {
    k_featT<64, 64><<<cdiv(BN, TPB), TPB, 0, stream>>>(feat, xtbuf);
    k_xx_t<64><<<cdiv(BN, TPB), TPB, 0, stream>>>(xtbuf, xxbuf);
    k_pd_t<64><<<dim3(NPTS / 64, NPTS / 16, BATCH), blk, 0, stream>>>(xtbuf, xxbuf, pdbuf);
    k_topk<<<dim3(NPTS / 4, BATCH), blk, 0, stream>>>(pdbuf, idxbuf, kk);
  };

  // ---- input transpose + knn on raw xyz ----
  k_transpose<<<cdiv(BN, TPB), TPB, 0, stream>>>(pc, xyzT);
  knn3(xyzT, KNN_K);

  // ---- transform net ----
  k_gconv6<<<dim3(NPTS * KNN_K / 64, 4, BATCH), blk, 0, stream>>>(
      xyzT, idxbuf, T_c1_w, T_c1_bias, T_bn1_g, T_bn1_b, T_bn1_rm, T_bn1_rv, big1,
      64, KNN_K);
  k_gemm_maxk_t<64><<<dim3(NPTS / 64, 8, BATCH), blk, 0, stream>>>(
      big1, T_c2_w, T_c2_bias, T_bn2_g, T_bn2_b, T_bn2_rm, T_bn2_rv, h128, 128, KNN_K);
  k_gemm_t<128><<<dim3(NPTS / 64, 64, BATCH), blk, 0, stream>>>(
      h128, T_c3_w, T_c3_bias, T_bn3_g, T_bn3_b, T_bn3_rm, T_bn3_rv, pdbuf,
      1024, NPTS, sX128, sY1024, 3);
  k_rmaxN<<<cdiv(BATCH * 1024, TPB), TPB, 0, stream>>>(pdbuf, pooled, 1024);
  k_gemm_t<1024><<<dim3(1, 32, 1), blk, 0, stream>>>(pooled, T_l1_w, T_l1_bias,
      T_bnl1_g, T_bnl1_b, T_bnl1_rm, T_bnl1_rv, l512, 512, BATCH, 0, 0, 3);
  k_gemm_t<512><<<dim3(1, 16, 1), blk, 0, stream>>>(l512, T_l2_w, T_l2_bias,
      T_bnl2_g, T_bnl2_b, T_bnl2_rm, T_bnl2_rv, l256, 256, BATCH, 0, 0, 3);
  k_gemm_t<256><<<dim3(1, 1, 1), blk, 0, stream>>>(l256, T_lt_w, T_lt_bias, nullptr,
      nullptr, nullptr, nullptr, q4, 4, BATCH, 0, 0, 0);
  k_quat<<<1, 32, 0, stream>>>(q4, out_tr, Rbuf);
  k_rotate<<<cdiv(BN, TPB), TPB, 0, stream>>>(xyzT, Rbuf, xyzrot, out_xyz);

  // ---- knn on rotated xyz (shared by shift net and x1 edge conv) ----
  knn3(xyzrot, KNN_K);

  // ---- shift net ----
  k_gconv6<<<dim3(NPTS * KNN_K / 64, 4, BATCH), blk, 0, stream>>>(
      xyzrot, idxbuf, S_c1_w, S_c1_bias, S_bn1_g, S_bn1_b, S_bn1_rm, S_bn1_rv, big1,
      64, KNN_K);
  k_gemm_maxk_t<64><<<dim3(NPTS / 64, 8, BATCH), blk, 0, stream>>>(
      big1, S_c2_w, S_c2_bias, S_bn2_g, S_bn2_b, S_bn2_rm, S_bn2_rv, h128, 128, KNN_K);
  k_gemm_t<128><<<dim3(NPTS / 64, 16, BATCH), blk, 0, stream>>>(
      h128, S_c3_w, S_c3_bias, S_bn3_g, S_bn3_b, S_bn3_rm, S_bn3_rv, f256,
      256, NPTS, sX128, sX256, 3);
  k_rmaxN<<<cdiv(BATCH * 256, TPB), TPB, 0, stream>>>(f256, g256, 256);
  k_concat<<<cdiv(BATCH * 512 * NPTS, TPB), TPB, 0, stream>>>(f256, g256, pdbuf);
  k_gemm_t<512><<<dim3(NPTS / 64, 8, BATCH), blk, 0, stream>>>(
      pdbuf, S_d1_w, S_d1_bias, S_bnd1_g, S_bnd1_b, S_bnd1_rm, S_bnd1_rv, h128,
      128, NPTS, sX512, sX128, 3);
  k_gemm_t<128><<<dim3(NPTS / 64, 4, BATCH), blk, 0, stream>>>(
      h128, S_d2_w, S_d2_bias, S_bnd2_g, S_bnd2_b, S_bnd2_rm, S_bnd2_rv, h64,
      64, NPTS, sX128, sX64, 3);
  k_gemm_t<64><<<dim3(NPTS / 64, 1, BATCH), blk, 0, stream>>>(
      h64, S_d3_w, S_d3_bias, nullptr, nullptr, nullptr, nullptr, out_off,
      3, NPTS, sX64, (long)3 * NPTS, 0);

  // ---- edge conv stack ----
  k_edgeconv_t<3><<<dim3(NPTS / 64, 4, BATCH), blk, 0, stream>>>(
      xyzrot, idxbuf, M_c1_w, M_c1_bias, M_bn1_g, M_bn1_b, M_bn1_rm, M_bn1_rv,
      nullptr, x1buf, 64, KNN_K, 1);
  knn64(x1buf, KNN_KMX);
  k_edgeconv_t<64><<<dim3(NPTS / 64, 4, BATCH), blk, 0, stream>>>(
      x1buf, idxbuf, M_c2_w, M_c2_bias, M_bn2_g, M_bn2_b, M_bn2_rm, M_bn2_rv,
      x1buf, out_x2, 64, KNN_K, 4);
  knn64(out_x2, KNN_K);
  k_edgeconv_t<64><<<dim3(NPTS / 64, 4, BATCH), blk, 0, stream>>>(
      out_x2, idxbuf, M_c3_w, M_c3_bias, M_bn3_g, M_bn3_b, M_bn3_rm, M_bn3_rv,
      out_x2, out_x3, 64, KNN_K, 1);
  knn64(out_x3, KNN_KMX);
  k_edgeconv_t<64><<<dim3(NPTS / 64, 4, BATCH), blk, 0, stream>>>(
      out_x3, idxbuf, M_c4_w, M_c4_bias, M_bn4_g, M_bn4_b, M_bn4_rm, M_bn4_rv,
      out_x3, out_x4, 64, KNN_K, 4);
}
